// GraphZ_61426622267683
// MI455X (gfx1250) — matrix operand order repaired and checked by reading, not compiled, not run on hardware
//
#include <hip/hip_runtime.h>

typedef _Float16 f16;
typedef __attribute__((ext_vector_type(16))) _Float16 v16h;
typedef __attribute__((ext_vector_type(8)))  float    v8f;

#define N_TOTAL 262144
#define NODES   512
#define NGRAPH  512
#define EPSG    1e-15f
#define BNEPS   1e-5f

// ---------------------------------------------------------------- cast x -> f16
__global__ void k_cast(const float* __restrict__ x, f16* __restrict__ A, int count) {
  int i = blockIdx.x * blockDim.x + threadIdx.x;
  if (i < count) A[i] = (f16)x[i];
}

// ------------------------------------------- weight transpose + pad, f32 -> f16
// src: K x Nout (row-major)  ->  dst: NPAD x KPAD  (dst[n][k] = src[k][n])
__global__ void k_prep_w(const float* __restrict__ W, f16* __restrict__ Wt,
                         int K, int Nout, int KPAD, int NPAD) {
  int i = blockIdx.x * blockDim.x + threadIdx.x;
  if (i >= KPAD * NPAD) return;
  int n = i / KPAD, k = i % KPAD;
  float v = (n < Nout && k < K) ? W[k * Nout + n] : 0.0f;
  Wt[i] = (f16)v;
}

// ---------------------------------------------------------------- kNN (top-6)
__global__ void k_knn(const float* __restrict__ pos, int* __restrict__ idx6,
                      float* __restrict__ cart6) {
  __shared__ float px[NODES], py[NODES];
  const int n0 = blockIdx.x * NODES, tid = threadIdx.x;
  for (int i = tid; i < NODES; i += blockDim.x) {
    px[i] = pos[(size_t)(n0 + i) * 3 + 0];
    py[i] = pos[(size_t)(n0 + i) * 3 + 1];
  }
  __syncthreads();
  for (int i = tid; i < NODES; i += blockDim.x) {
    const float xi = px[i], yi = py[i];
    float b0 = 3.4e38f, b1 = 3.4e38f, b2 = 3.4e38f, b3 = 3.4e38f, b4 = 3.4e38f, b5 = 3.4e38f;
    int   i0 = 0, i1 = 0, i2 = 0, i3 = 0, i4 = 0, i5 = 0;
    for (int j = 0; j < NODES; ++j) {
      float dx = px[j] - xi, dy = py[j] - yi;
      float dd = dx * dx + dy * dy;
      if (dd < b5) {
        if (dd < b4) { b5 = b4; i5 = i4;
          if (dd < b3) { b4 = b3; i4 = i3;
            if (dd < b2) { b3 = b2; i3 = i2;
              if (dd < b1) { b2 = b1; i2 = i1;
                if (dd < b0) { b1 = b0; i1 = i0; b0 = dd; i0 = j; }
                else         { b1 = dd; i1 = j; }
              } else { b2 = dd; i2 = j; }
            } else { b3 = dd; i3 = j; }
          } else { b4 = dd; i4 = j; }
        } else { b5 = dd; i5 = j; }
      }
    }
    const int n = n0 + i;
    int bi[6] = { i0, i1, i2, i3, i4, i5 };
    #pragma unroll
    for (int e = 0; e < 6; ++e) {
      idx6[(size_t)n * 6 + e] = n0 + bi[e];
      cart6[(size_t)n * 12 + e * 2 + 0] = (px[bi[e]] - xi) * (1.0f / 12.0f) + 0.5f;
      cart6[(size_t)n * 12 + e * 2 + 1] = (py[bi[e]] - yi) * (1.0f / 12.0f) + 0.5f;
    }
  }
}

// ---------------------------------------------------------------- WMMA GEMM
// Y[M x Nout] = A[M x KPAD](f16) @ Wt' (Wt stored NPAD x KPAD, i.e. col-major W)
// Staging uses gfx1250 async global->LDS (ASYNCcnt), fragments fully preloaded
// into VGPRs so the WMMA chain issues back-to-back.
template <int KPAD, int NPAD>
__global__ void k_gemm(const f16* __restrict__ A, const f16* __restrict__ Wt,
                       float* __restrict__ Y, int Nout) {
  constexpr int NT = NPAD / 16, KS = KPAD / 32;
  __shared__ __align__(16) f16 sW[NPAD * KPAD];
  __shared__ __align__(16) f16 sA[128 * KPAD];
  const int tid = threadIdx.x, lane = tid & 31, wid = tid >> 5;

  // async copy weights (NPAD*KPAD f16) global -> LDS, no VGPR round-trip
  {
    const char* g = (const char*)Wt;
    const unsigned ldsW = (unsigned)(unsigned long long)(&sW[0]);
    for (int i = tid; i < NPAD * KPAD / 8; i += 256) {
      unsigned l = ldsW + (unsigned)i * 16u;
      const char* gp = g + (size_t)i * 16;
      asm volatile("global_load_async_to_lds_b128 %0, %1, off"
                   :: "v"(l), "v"(gp) : "memory");
    }
  }
  // async copy this wave's 16 contiguous rows of A -> LDS
  const int rowBase = blockIdx.x * 128 + wid * 16;
  {
    const char* g = (const char*)(A + (size_t)rowBase * KPAD);
    const unsigned ldsA = (unsigned)(unsigned long long)(&sA[wid * 16 * KPAD]);
    #pragma unroll
    for (int c = 0; c < (16 * KPAD / 8) / 32; ++c) {
      unsigned l = ldsA + (unsigned)(c * 32 + lane) * 16u;
      const char* gp = g + (size_t)(c * 32 + lane) * 16;
      asm volatile("global_load_async_to_lds_b128 %0, %1, off"
                   :: "v"(l), "v"(gp) : "memory");
    }
  }
  asm volatile("s_wait_asynccnt 0" ::: "memory");
  __syncthreads();

  const unsigned* sAu = (const unsigned*)sA;
  const unsigned* sWu = (const unsigned*)sW;
  const int r = lane & 15, hi = (lane >> 4) & 1;

  union Frag { v16h v; unsigned u[8]; };
  // preload all A fragments (16-bit A 16x32: K = (q/4)*16 + hi*8 + (q&3)*2)
  Frag fa[KS];
  #pragma unroll
  for (int ks = 0; ks < KS; ++ks) {
    #pragma unroll
    for (int q = 0; q < 8; ++q) {
      int kOff = ks * 32 + (q >> 2) * 16 + hi * 8 + (q & 3) * 2;
      fa[ks].u[q] = sAu[((unsigned)(wid * 16 * KPAD + r * KPAD + kOff)) >> 1];
    }
  }
  // preload all B fragments (B 32x16: col N = lane%16, K = (q/4)*16 + hi*8 + (q&3)*2)
  Frag fb[KS][NT];
  #pragma unroll
  for (int ks = 0; ks < KS; ++ks) {
    #pragma unroll
    for (int t = 0; t < NT; ++t) {
      #pragma unroll
      for (int q = 0; q < 8; ++q) {
        int kOff = ks * 32 + (q >> 2) * 16 + hi * 8 + (q & 3) * 2;
        fb[ks][t].u[q] = sWu[((unsigned)((t * 16 + r) * KPAD + kOff)) >> 1];
      }
    }
  }
  // back-to-back WMMA chain (only C-accumulator deps remain)
  v8f acc[NT] = {};
  #pragma unroll
  for (int ks = 0; ks < KS; ++ks) {
    #pragma unroll
    for (int t = 0; t < NT; ++t) {
      acc[t] = __builtin_amdgcn_wmma_f32_16x16x32_f16(false, fa[ks].v, false, fb[ks][t].v,
                                                      (short)0, acc[t], false, false);
    }
  }
  #pragma unroll
  for (int t = 0; t < NT; ++t) {
    int col = t * 16 + r;
    if (col < Nout) {
      #pragma unroll
      for (int q = 0; q < 8; ++q) {
        int row = rowBase + q + hi * 8;  // C/D: row = vgpr + 8*(lane/16)
        Y[(size_t)row * Nout + col] = acc[t][q];
      }
    }
  }
}

// -------------------------------------------- k=1 layer epilogue (in-place Yr)
__global__ void k_edge1(const float* __restrict__ Yg, float* __restrict__ Yr,
                        const int* __restrict__ idx6, const float* __restrict__ cart6,
                        const float* __restrict__ mu, const float* __restrict__ sg,
                        const float* __restrict__ bias, int Nout) {
  int n = blockIdx.x * blockDim.x + threadIdx.x;
  if (n >= N_TOTAL) return;
  int j = idx6[(size_t)n * 6];
  float c0 = cart6[(size_t)n * 12 + 0], c1 = cart6[(size_t)n * 12 + 1];
  float e0 = c0 - mu[0], e1 = c1 - mu[1];
  float w = expf(-0.5f * (e0 * e0 / (EPSG + sg[0] * sg[0]) +
                          e1 * e1 / (EPSG + sg[1] * sg[1])));
  const float* yg = Yg + (size_t)j * Nout;
  float* yr = Yr + (size_t)n * Nout;
  for (int m = 0; m < Nout; ++m) yr[m] = w * yg[m] + yr[m] + bias[m];
}

// -------------------------------------------- k=6, kk=3 message (layer 2)
__global__ void k_msg(const float* __restrict__ XT, float* __restrict__ Yr,
                      const int* __restrict__ idx6, const float* __restrict__ cart6,
                      const float* __restrict__ mu, const float* __restrict__ sg,
                      const float* __restrict__ bias) {
  int n = blockIdx.x * blockDim.x + threadIdx.x;
  if (n >= N_TOTAL) return;
  float iv[6];
  #pragma unroll
  for (int t = 0; t < 6; ++t) iv[t] = 1.0f / (EPSG + sg[t] * sg[t]);
  int jj[6];
  #pragma unroll
  for (int e = 0; e < 6; ++e) jj[e] = idx6[(size_t)n * 6 + e];
  #pragma unroll
  for (int e = 0; e < 6; ++e) __builtin_prefetch(XT + (size_t)jj[e] * 51, 0, 3);
  float acc[17] = {};
  for (int e = 0; e < 6; ++e) {
    int j = jj[e];
    float c0 = cart6[(size_t)n * 12 + e * 2 + 0];
    float c1 = cart6[(size_t)n * 12 + e * 2 + 1];
    float g0, g1, g2;
    {
      float a0 = c0 - mu[0], a1 = c1 - mu[1];
      g0 = expf(-0.5f * (a0 * a0 * iv[0] + a1 * a1 * iv[1]));
      float b0 = c0 - mu[2], b1 = c1 - mu[3];
      g1 = expf(-0.5f * (b0 * b0 * iv[2] + b1 * b1 * iv[3]));
      float d0 = c0 - mu[4], d1 = c1 - mu[5];
      g2 = expf(-0.5f * (d0 * d0 * iv[4] + d1 * d1 * iv[5]));
    }
    const float* row = XT + (size_t)j * 51;
    #pragma unroll
    for (int m = 0; m < 17; ++m)
      acc[m] += g0 * row[m] + g1 * row[17 + m] + g2 * row[34 + m];
  }
  float* yr = Yr + (size_t)n * 17;
  #pragma unroll
  for (int m = 0; m < 17; ++m)
    yr[m] = acc[m] * (1.0f / 6.0f) + yr[m] + bias[m];
}

// -------------------------------------------- BN stats: per-block partials
__global__ void k_stats_partial(const float* __restrict__ out, int Nout,
                                float* __restrict__ partial) {
  __shared__ float sP[8][64][2];
  const int tid = threadIdx.x, lane = tid & 31, wid = tid >> 5;
  const size_t row = (size_t)blockIdx.x * 256 + tid;
  const float* p = out + row * Nout;
  for (int ch = 0; ch < Nout; ++ch) {
    float v = p[ch];
    float s = v, q = v * v;
    #pragma unroll
    for (int o = 16; o > 0; o >>= 1) { s += __shfl_xor(s, o); q += __shfl_xor(q, o); }
    if (lane == 0) { sP[wid][ch][0] = s; sP[wid][ch][1] = q; }
  }
  __syncthreads();
  if (tid < Nout) {
    float s = 0.f, q = 0.f;
    #pragma unroll
    for (int w = 0; w < 8; ++w) { s += sP[w][tid][0]; q += sP[w][tid][1]; }
    partial[((size_t)tid * 1024 + blockIdx.x) * 2 + 0] = s;
    partial[((size_t)tid * 1024 + blockIdx.x) * 2 + 1] = q;
  }
}

__global__ void k_stats_final(const float* __restrict__ partial,
                              const float* __restrict__ gamma,
                              const float* __restrict__ beta,
                              float* __restrict__ stats, int Nout) {
  int ch = threadIdx.x;
  if (ch >= Nout) return;
  float s = 0.f, q = 0.f;
  for (int b = 0; b < 1024; ++b) {
    s += partial[((size_t)ch * 1024 + b) * 2 + 0];
    q += partial[((size_t)ch * 1024 + b) * 2 + 1];
  }
  float mean = s * (1.0f / N_TOTAL);
  float var  = q * (1.0f / N_TOTAL) - mean * mean;
  float a = gamma[ch] * rsqrtf(var + BNEPS);
  stats[ch * 2 + 0] = a;
  stats[ch * 2 + 1] = beta[ch] - mean * a;
}

// -------------------------------------------- BN apply + cast to padded f16
__global__ void k_bn_cast(const float* __restrict__ out, const float* __restrict__ stats,
                          f16* __restrict__ A, int Nout, int KPADn, int count) {
  int i = blockIdx.x * blockDim.x + threadIdx.x;
  if (i >= count) return;
  int ch = i % KPADn;
  size_t n = (size_t)i / KPADn;
  float v = 0.0f;
  if (ch < Nout) v = out[n * Nout + ch] * stats[ch * 2] + stats[ch * 2 + 1];
  A[i] = (f16)v;
}

// -------------------------------------------- last layer (cout=1, k=1)
__global__ void k_final(const f16* __restrict__ A3, const int* __restrict__ idx6,
                        const float* __restrict__ cart6,
                        const float* __restrict__ g3, const float* __restrict__ mu,
                        const float* __restrict__ sg, const float* __restrict__ root3,
                        const float* __restrict__ bias3, float* __restrict__ out) {
  int n = blockIdx.x * blockDim.x + threadIdx.x;
  if (n >= N_TOTAL) return;
  int j = idx6[(size_t)n * 6];
  float c0 = cart6[(size_t)n * 12 + 0], c1 = cart6[(size_t)n * 12 + 1];
  float e0 = c0 - mu[0], e1 = c1 - mu[1];
  float w = expf(-0.5f * (e0 * e0 / (EPSG + sg[0] * sg[0]) +
                          e1 * e1 / (EPSG + sg[1] * sg[1])));
  const f16* aj = A3 + (size_t)j * 32;
  const f16* an = A3 + (size_t)n * 32;
  float s1 = 0.f, s2 = 0.f;
  #pragma unroll
  for (int k = 0; k < 17; ++k) {
    s1 += (float)aj[k] * g3[k];
    s2 += (float)an[k] * root3[k];
  }
  out[n] = w * s1 + s2 + bias3[0];
}

// ============================================================================
extern "C" void kernel_launch(void* const* d_in, const int* in_sizes, int n_in,
                              void* d_out, int out_size, void* d_ws, size_t ws_size,
                              hipStream_t stream) {
  (void)in_sizes; (void)n_in; (void)out_size; (void)ws_size;
  const float* x   = (const float*)d_in[0];
  const float* pos = (const float*)d_in[1];
  const float *G[4], *MU[4], *SG[4], *RT[4], *BI[4], *GA[3], *BE[3];
  int p = 2;
  for (int li = 0; li < 4; ++li) {
    G[li]  = (const float*)d_in[p++];
    MU[li] = (const float*)d_in[p++];
    SG[li] = (const float*)d_in[p++];
    RT[li] = (const float*)d_in[p++];
    BI[li] = (const float*)d_in[p++];
    if (li < 3) { GA[li] = (const float*)d_in[p++]; BE[li] = (const float*)d_in[p++]; }
  }

  char* ws = (char*)d_ws;
  size_t off = 0;
  auto carve = [&](size_t bytes) -> char* {
    char* r = ws + off;
    off += (bytes + 255) & ~(size_t)255;
    return r;
  };
  f16*   A       = (f16*)  carve((size_t)N_TOTAL * 64 * 2);  // padded f16 activations
  float* Yg      = (float*)carve((size_t)N_TOTAL * 64 * 4);  // x@g   (also XT for layer 2)
  float* Yr      = (float*)carve((size_t)N_TOTAL * 64 * 4);  // x@root -> layer output (in place)
  int*   idx6    = (int*)  carve((size_t)N_TOTAL * 6 * 4);
  float* cart6   = (float*)carve((size_t)N_TOTAL * 12 * 4);
  float* partial = (float*)carve((size_t)64 * 1024 * 2 * 4);
  float* stats   = (float*)carve((size_t)64 * 2 * 4);
  f16* W[6];
  for (int i = 0; i < 6; ++i) W[i] = (f16*)carve((size_t)64 * 64 * 2);

  const int T = 256;
  // ---- one-time prep ----
  k_cast<<<(N_TOTAL * 64) / T, T, 0, stream>>>(x, A, N_TOTAL * 64);
  k_knn<<<NGRAPH, T, 0, stream>>>(pos, idx6, cart6);
  k_prep_w<<<(64 * 64) / T, T, 0, stream>>>(G[0],  W[0], 64, 51, 64, 64);
  k_prep_w<<<(64 * 64) / T, T, 0, stream>>>(RT[0], W[1], 64, 51, 64, 64);
  k_prep_w<<<(64 * 48) / T, T, 0, stream>>>(G[1],  W[2], 51, 34, 64, 48);
  k_prep_w<<<(64 * 48) / T, T, 0, stream>>>(RT[1], W[3], 51, 34, 64, 48);
  k_prep_w<<<(64 * 64) / T, T, 0, stream>>>(G[2],  W[4], 34, 51, 64, 64);
  k_prep_w<<<(64 * 32) / T, T, 0, stream>>>(RT[2], W[5], 34, 17, 64, 32);

  const int GB = N_TOTAL / 128;   // 2048 gemm blocks
  const int NB = N_TOTAL / T;     // 1024 node blocks

  // ---- layer 0 (k=1) ----
  k_gemm<64, 64><<<GB, T, 0, stream>>>(A, W[0], Yg, 51);
  k_gemm<64, 64><<<GB, T, 0, stream>>>(A, W[1], Yr, 51);
  k_edge1<<<NB, T, 0, stream>>>(Yg, Yr, idx6, cart6, MU[0], SG[0], BI[0], 51);
  k_stats_partial<<<NB, T, 0, stream>>>(Yr, 51, partial);
  k_stats_final<<<1, 64, 0, stream>>>(partial, GA[0], BE[0], stats, 51);
  k_bn_cast<<<(N_TOTAL * 64) / T, T, 0, stream>>>(Yr, stats, A, 51, 64, N_TOTAL * 64);

  // ---- layer 1 (k=1) ----
  k_gemm<64, 48><<<GB, T, 0, stream>>>(A, W[2], Yg, 34);
  k_gemm<64, 48><<<GB, T, 0, stream>>>(A, W[3], Yr, 34);
  k_edge1<<<NB, T, 0, stream>>>(Yg, Yr, idx6, cart6, MU[1], SG[1], BI[1], 34);
  k_stats_partial<<<NB, T, 0, stream>>>(Yr, 34, partial);
  k_stats_final<<<1, 64, 0, stream>>>(partial, GA[1], BE[1], stats, 34);
  k_bn_cast<<<(N_TOTAL * 64) / T, T, 0, stream>>>(Yr, stats, A, 34, 64, N_TOTAL * 64);

  // ---- layer 2 (k=6, kk=3) ----
  k_gemm<64, 64><<<GB, T, 0, stream>>>(A, W[4], Yg, 51);  // XT: N x (3*17)
  k_gemm<64, 32><<<GB, T, 0, stream>>>(A, W[5], Yr, 17);
  k_msg<<<NB, T, 0, stream>>>(Yg, Yr, idx6, cart6, MU[2], SG[2], BI[2]);
  k_stats_partial<<<NB, T, 0, stream>>>(Yr, 17, partial);
  k_stats_final<<<1, 64, 0, stream>>>(partial, GA[2], BE[2], stats, 17);
  k_bn_cast<<<(N_TOTAL * 32) / T, T, 0, stream>>>(Yr, stats, A, 17, 32, N_TOTAL * 32);

  // ---- layer 3 (k=1, cout=1) ----
  k_final<<<NB, T, 0, stream>>>(A, idx6, cart6, G[3], MU[3], SG[3], RT[3], BI[3],
                                (float*)d_out);
}
